// TweetDisasterClassifierCNN_60687887892814
// MI455X (gfx1250) — matrix operand order repaired and checked by reading, not compiled, not run on hardware
//
#include <hip/hip_runtime.h>
#include <hip/hip_bf16.h>
#include <math.h>

// CDNA5 / gfx1250 TextCNN classifier.
// Strategy: conv-as-GEMM. dots[B*S, 64] = hidden[B*S,768] @ Wcat[768,64]
// via v_wmma_f32_16x16x32_f16 (fp32 accumulate), then tiny epilogue
// (compaction permutation + shifted sums + masked maxpool + FC + sigmoid).
// Memory bound: one pass over 402 MB of hidden -> ~17us floor at 23.3 TB/s.

typedef __attribute__((ext_vector_type(16))) _Float16 v16h;
typedef __attribute__((ext_vector_type(8)))  float    v8f;

#define D_DIM   768
#define S_DIM   128
#define NCOL    64      // 56 real columns, padded
#define KBLKS   24      // 768 / 32
#define NEGBIG  -1e30f

// ---------------------------------------------------------------------------
// Kernel 1: pack Wcat[768,56] (from w2..w5) into exact B-fragment layout:
//   bpack[kblk][ntile][lane][e]  (f16), e = 0..15
//   B layout (16-bit, 32x16): lanes0-15 hold K=0..7 & 16..23 (N=lane),
//                             lanes16-31 hold K=8..15 & 24..31 (N=lane-16).
// Column order c = concat over k in {2,3,4,5} of (f*k + j):
//   bases {0, 8, 20, 36}, total 56, cols 56..63 are zero.
// ---------------------------------------------------------------------------
__global__ void pack_weights_kernel(const float* __restrict__ w2,
                                    const float* __restrict__ w3,
                                    const float* __restrict__ w4,
                                    const float* __restrict__ w5,
                                    _Float16* __restrict__ bpack) {
    int idx = blockIdx.x * blockDim.x + threadIdx.x;
    const int TOTAL = KBLKS * 4 * 32 * 16;   // 49152 halves
    if (idx >= TOTAL) return;

    int e    =  idx        & 15;
    int lane = (idx >> 4)  & 31;
    int nt   = (idx >> 9)  & 3;
    int kblk =  idx >> 11;

    int n  = lane & 15;
    int c  = nt * 16 + n;                    // output column 0..63
    int kk = ((e >> 3) << 4) + ((lane >> 4) << 3) + (e & 7); // 0..31 within k-block
    int k  = kblk * 32 + kk;                 // reduction dim 0..767

    float val = 0.0f;
    if (c < 56) {
        int ki, ksz, cl;
        if      (c <  8) { ki = 0; ksz = 2; cl = c;      }
        else if (c < 20) { ki = 1; ksz = 3; cl = c - 8;  }
        else if (c < 36) { ki = 2; ksz = 4; cl = c - 20; }
        else             { ki = 3; ksz = 5; cl = c - 36; }
        int f = cl / ksz;
        int j = cl % ksz;
        const float* w = (ki == 0) ? w2 : (ki == 1) ? w3 : (ki == 2) ? w4 : w5;
        // w shape [F, D, ksz] row-major
        val = w[((size_t)f * D_DIM + k) * ksz + j];
    }
    bpack[idx] = (_Float16)val;
}

// ---------------------------------------------------------------------------
// Kernel 2: per-batch-row fused GEMM + epilogue. 1024 blocks x 256 threads.
// Wave w (of 8) computes dot rows m0..m0+15 for all 64 columns.
// ---------------------------------------------------------------------------
__global__ __launch_bounds__(256) void cnn_main_kernel(
    const int*      __restrict__ x,
    const float*    __restrict__ hidden,
    const _Float16* __restrict__ bpack,
    const float*    __restrict__ b2, const float* __restrict__ b3,
    const float*    __restrict__ b4, const float* __restrict__ b5,
    const float*    __restrict__ fc_w, const float* __restrict__ fc_b,
    float*          __restrict__ out)
{
    __shared__ float dots[S_DIM][NCOL + 1];   // +1 pad vs LDS bank conflicts
    __shared__ int   comp[S_DIM];
    __shared__ int   npf[S_DIM];
    __shared__ int   s_len;
    __shared__ float feats[16];

    const int b    = blockIdx.x;
    const int tid  = threadIdx.x;
    const int lane = tid & 31;
    const int wv   = tid >> 5;               // wave id 0..7 (wave32)

    // --- pad flags (epilogue input), overlapped with GEMM ---
    if (tid < S_DIM) npf[tid] = (x[(size_t)b * S_DIM + tid] != 0) ? 1 : 0;

    // --- GEMM: 16 rows x 64 cols per wave, K = 768 in 24 steps of 32 ---
    const int m0    = wv * 16;
    const int row   = lane & 15;
    const int khalf = lane >> 4;             // 0 or 1
    // A layout (16-bit 16x32): lane<16 -> K 0..7 & 16..23 ; lane>=16 -> +8
    const float* hrowbase =
        hidden + ((size_t)b * S_DIM + m0 + row) * D_DIM + khalf * 8;

    v8f acc0 = {}, acc1 = {}, acc2 = {}, acc3 = {};

    for (int kb = 0; kb < KBLKS; ++kb) {
        const float* hp = hrowbase + kb * 32;
        __builtin_prefetch(hp + 64, 0, 1);   // global_prefetch_b8 (stream hint)

        float4 a0 = *(const float4*)(hp);
        float4 a1 = *(const float4*)(hp + 4);
        float4 a2 = *(const float4*)(hp + 16);
        float4 a3 = *(const float4*)(hp + 20);

        v16h a;
        a[0]  = (_Float16)a0.x; a[1]  = (_Float16)a0.y;
        a[2]  = (_Float16)a0.z; a[3]  = (_Float16)a0.w;
        a[4]  = (_Float16)a1.x; a[5]  = (_Float16)a1.y;
        a[6]  = (_Float16)a1.z; a[7]  = (_Float16)a1.w;
        a[8]  = (_Float16)a2.x; a[9]  = (_Float16)a2.y;
        a[10] = (_Float16)a2.z; a[11] = (_Float16)a2.w;
        a[12] = (_Float16)a3.x; a[13] = (_Float16)a3.y;
        a[14] = (_Float16)a3.z; a[15] = (_Float16)a3.w;

        // B fragments: contiguous, fully coalesced (L2-resident weights)
        const v16h* bp = (const v16h*)bpack + ((size_t)kb * 4) * 32 + lane;
        v16h bf0 = bp[0];
        v16h bf1 = bp[32];
        v16h bf2 = bp[64];
        v16h bf3 = bp[96];

        acc0 = __builtin_amdgcn_wmma_f32_16x16x32_f16(
                   false, a, false, bf0, (short)0, acc0, false, false);
        acc1 = __builtin_amdgcn_wmma_f32_16x16x32_f16(
                   false, a, false, bf1, (short)0, acc1, false, false);
        acc2 = __builtin_amdgcn_wmma_f32_16x16x32_f16(
                   false, a, false, bf2, (short)0, acc2, false, false);
        acc3 = __builtin_amdgcn_wmma_f32_16x16x32_f16(
                   false, a, false, bf3, (short)0, acc3, false, false);
    }

    // --- C layout: VGPR r: lanes0-15 M=r, lanes16-31 M=8+r; N=lane&15 ---
    const int mbase = m0 + (khalf << 3);
    #pragma unroll
    for (int r = 0; r < 8; ++r) {
        dots[mbase + r][ 0 + row] = acc0[r];
        dots[mbase + r][16 + row] = acc1[r];
        dots[mbase + r][32 + row] = acc2[r];
        dots[mbase + r][48 + row] = acc3[r];
    }
    __syncthreads();

    // --- stable compaction: comp[i] = original index of i-th non-pad ---
    if (tid < S_DIM && npf[tid]) {
        int r = 0;
        for (int i = 0; i < tid; ++i) r += npf[i];
        comp[r] = tid;
    }
    if (tid == 0) {
        int n = 0;
        for (int i = 0; i < S_DIM; ++i) n += npf[i];
        s_len = n;
    }
    __syncthreads();

    // --- conv (shifted dot sums) + masked maxpool: one thread per (k,f) ---
    if (tid < 16) {
        const int ki  = tid >> 2;            // 0..3  -> k = 2..5
        const int f   = tid & 3;
        const int ksz = ki + 2;
        const int basec[4] = {0, 8, 20, 36};
        const float* bias_p = (ki == 0) ? b2 : (ki == 1) ? b3
                            : (ki == 2) ? b4 : b5;
        const float bias = bias_p[f];
        const int   cb   = basec[ki] + f * ksz;

        float mx = NEGBIG;
        const int tmax = s_len - ksz + 1;    // may be <= 0 -> feature = NEG
        for (int t = 0; t < tmax; ++t) {
            float v = bias;
            #pragma unroll 5
            for (int j = 0; j < ksz; ++j)
                v += dots[comp[t + j]][cb + j];
            mx = fmaxf(mx, v);
        }
        feats[tid] = mx;                     // order ki*F + f matches concat
    }
    __syncthreads();

    // --- FC + sigmoid ---
    if (tid == 0) {
        float z = fc_b[0];
        #pragma unroll
        for (int i = 0; i < 16; ++i) z += feats[i] * fc_w[i];
        out[b] = 1.0f / (1.0f + __expf(-z));
    }
}

// ---------------------------------------------------------------------------
extern "C" void kernel_launch(void* const* d_in, const int* in_sizes, int n_in,
                              void* d_out, int out_size, void* d_ws, size_t ws_size,
                              hipStream_t stream) {
    const int*   x      = (const int*)  d_in[0];
    const float* hidden = (const float*)d_in[1];
    const float* w2     = (const float*)d_in[2];
    const float* b2     = (const float*)d_in[3];
    const float* w3     = (const float*)d_in[4];
    const float* b3     = (const float*)d_in[5];
    const float* w4     = (const float*)d_in[6];
    const float* b4     = (const float*)d_in[7];
    const float* w5     = (const float*)d_in[8];
    const float* b5     = (const float*)d_in[9];
    const float* fc_w   = (const float*)d_in[10];
    const float* fc_b   = (const float*)d_in[11];
    float*       out    = (float*)d_out;

    _Float16* bpack = (_Float16*)d_ws;       // 24*4*32*16 halves = 96 KB

    const int PACK_TOTAL = KBLKS * 4 * 32 * 16;
    pack_weights_kernel<<<(PACK_TOTAL + 255) / 256, 256, 0, stream>>>(
        w2, w3, w4, w5, bpack);

    cnn_main_kernel<<<1024, 256, 0, stream>>>(
        x, hidden, bpack, b2, b3, b4, b5, fc_w, fc_b, out);
}
